// SelfAttention_40759239639086
// MI455X (gfx1250) — matrix operand order repaired and checked by reading, not compiled, not run on hardware
//
#include <hip/hip_runtime.h>

typedef __attribute__((ext_vector_type(16))) _Float16 v16h;
typedef __attribute__((ext_vector_type(8)))  float    v8f;

#define KB  8
#define KC  256
#define KC8 32
#define KN  4096

union U16 { v16h h; unsigned int u[8]; };

// ---------------------------------------------------------------------------
// Kernel 0: fold depthwise affine into pointwise weights, convert to f16.
//   W'[o,c] = W[o,c] * dw_w[c] ; b'[o] = pw_b[o] + sum_c W[o,c]*dw_b[c]
// ---------------------------------------------------------------------------
__global__ void fold_weights_kernel(
    const float* __restrict__ q_dw_w, const float* __restrict__ q_dw_b,
    const float* __restrict__ q_pw_w, const float* __restrict__ q_pw_b,
    const float* __restrict__ k_dw_w, const float* __restrict__ k_dw_b,
    const float* __restrict__ k_pw_w, const float* __restrict__ k_pw_b,
    const float* __restrict__ v_dw_w, const float* __restrict__ v_dw_b,
    const float* __restrict__ v_pw_w, const float* __restrict__ v_pw_b,
    _Float16* __restrict__ wq16, _Float16* __restrict__ wk16,
    _Float16* __restrict__ wv16,
    float* __restrict__ bq, float* __restrict__ bk, float* __restrict__ bv)
{
  __shared__ float red[KC];
  int o = blockIdx.x, t = threadIdx.x;
  const float *pw, *dww, *dwb, *pb; _Float16* dst; float* bdst; int oo;
  if (o < KC8)          { oo = o;          pw = q_pw_w; dww = q_dw_w; dwb = q_dw_b; pb = q_pw_b; dst = wq16; bdst = bq; }
  else if (o < 2 * KC8) { oo = o - KC8;    pw = k_pw_w; dww = k_dw_w; dwb = k_dw_b; pb = k_pw_b; dst = wk16; bdst = bk; }
  else                  { oo = o - 2*KC8;  pw = v_pw_w; dww = v_dw_w; dwb = v_dw_b; pb = v_pw_b; dst = wv16; bdst = bv; }
  float w = pw[oo * KC + t];
  dst[oo * KC + t] = (_Float16)(w * dww[t]);
  red[t] = w * dwb[t];
  __syncthreads();
  for (int s = KC / 2; s > 0; s >>= 1) { if (t < s) red[t] += red[t + s]; __syncthreads(); }
  if (t == 0) bdst[oo] = pb[oo] + red[0];
}

// ---------------------------------------------------------------------------
// Kernel 1: q/k/v projections via WMMA. One wave handles a 16-pixel tile.
//   qT/kT layout: [b][n][c]  (f16)   v layout: [b][c][n]  (f16)
// ---------------------------------------------------------------------------
__global__ void __launch_bounds__(32) proj_qkv_kernel(
    const float* __restrict__ x,
    const _Float16* __restrict__ wq16, const _Float16* __restrict__ wk16,
    const _Float16* __restrict__ wv16,
    const float* __restrict__ bq, const float* __restrict__ bk,
    const float* __restrict__ bv,
    _Float16* __restrict__ qT, _Float16* __restrict__ kT,
    _Float16* __restrict__ vK)
{
  __shared__ _Float16 ylds[16][KC];   // [pixel][channel] so K-pairs are contiguous
  const int lane = threadIdx.x;
  const int hi = lane >> 4, ln = lane & 15;
  const int b  = blockIdx.x >> 8;
  const int i0 = (blockIdx.x & 255) << 4;
  const float* xb = x + (size_t)b * KC * KN;

  for (int idx = lane; idx < KC * 16; idx += 32) {
    int c = idx >> 4, i = idx & 15;
    ylds[i][c] = (_Float16)xb[(size_t)c * KN + i0 + i];
  }
  __syncthreads();

  const unsigned int* yrow = (const unsigned int*)&ylds[ln][0];

  // ---- pass 1: q and k (two 16-row o-tiles each) ----
  v8f qa[2], ka[2];
#pragma unroll
  for (int ot = 0; ot < 2; ++ot)
#pragma unroll
    for (int r = 0; r < 8; ++r) {
      qa[ot][r] = bq[ot * 16 + r + 8 * hi];
      ka[ot][r] = bk[ot * 16 + r + 8 * hi];
    }
#pragma unroll 1
  for (int kc = 0; kc < KC; kc += 32) {
    U16 yb;
#pragma unroll
    for (int v = 0; v < 8; ++v) yb.u[v] = yrow[(kc + hi * 16 + 2 * v) >> 1];  // B layout
#pragma unroll
    for (int ot = 0; ot < 2; ++ot) {
      U16 aq, ak;
      const unsigned int* qwrow = (const unsigned int*)(wq16 + (size_t)(ot * 16 + ln) * KC);
      const unsigned int* kwrow = (const unsigned int*)(wk16 + (size_t)(ot * 16 + ln) * KC);
#pragma unroll
      for (int v = 0; v < 8; ++v) {
        int K = kc + hi * 16 + 2 * v;          // A layout
        aq.u[v] = qwrow[K >> 1];
        ak.u[v] = kwrow[K >> 1];
      }
      qa[ot] = __builtin_amdgcn_wmma_f32_16x16x32_f16(false, aq.h, false, yb.h, (short)0, qa[ot], false, false);
      ka[ot] = __builtin_amdgcn_wmma_f32_16x16x32_f16(false, ak.h, false, yb.h, (short)0, ka[ot], false, false);
    }
  }
  {
    _Float16* qdst = qT + ((size_t)b * KN + i0 + ln) * KC8;
    _Float16* kdst = kT + ((size_t)b * KN + i0 + ln) * KC8;
#pragma unroll
    for (int ot = 0; ot < 2; ++ot)
#pragma unroll
      for (int r = 0; r < 8; ++r) {
        int o = ot * 16 + r + 8 * hi;        // C layout: row = r + 8*hi, col = ln
        qdst[o] = (_Float16)qa[ot][r];
        kdst[o] = (_Float16)ka[ot][r];
      }
  }

  // ---- pass 2: v (sixteen 16-row o-tiles) ----
  v8f va[16];
#pragma unroll
  for (int ot = 0; ot < 16; ++ot)
#pragma unroll
    for (int r = 0; r < 8; ++r) va[ot][r] = bv[ot * 16 + r + 8 * hi];
#pragma unroll 1
  for (int kc = 0; kc < KC; kc += 32) {
    U16 yb;
#pragma unroll
    for (int v = 0; v < 8; ++v) yb.u[v] = yrow[(kc + hi * 16 + 2 * v) >> 1];
#pragma unroll
    for (int ot = 0; ot < 16; ++ot) {
      U16 av;
      const unsigned int* wrow = (const unsigned int*)(wv16 + (size_t)(ot * 16 + ln) * KC);
#pragma unroll
      for (int v = 0; v < 8; ++v) {
        int K = kc + hi * 16 + 2 * v;
        av.u[v] = wrow[K >> 1];
      }
      va[ot] = __builtin_amdgcn_wmma_f32_16x16x32_f16(false, av.h, false, yb.h, (short)0, va[ot], false, false);
    }
  }
  {
    _Float16* vdst = vK + (size_t)b * KC * KN + i0 + ln;
#pragma unroll
    for (int ot = 0; ot < 16; ++ot)
#pragma unroll
      for (int r = 0; r < 8; ++r) {
        int o = ot * 16 + r + 8 * hi;
        vdst[(size_t)o * KN] = (_Float16)va[ot][r];
      }
  }
}

// ---------------------------------------------------------------------------
// Kernel 2: flash attention, transposed-score formulation.
//   S^T[j,i] = sum_c k[c,j] q[c,i]   (A = k-tile, B = q-tile)
//   O^T[c,i] += sum_n v[c,n] P^T[n,i] (A = v-tile, B = P^T built via shfl)
// One wave per (batch, 16-query tile, 128-channel half). 128 keys per step.
// Softmax stats are per-lane scalars; P transpose needs only xor-16 exchange.
// ---------------------------------------------------------------------------
__global__ void __launch_bounds__(32) attn_kernel(
    const float* __restrict__ x,
    const _Float16* __restrict__ qT, const _Float16* __restrict__ kT,
    const _Float16* __restrict__ vK,
    const float* __restrict__ gamma, float* __restrict__ out)
{
  const int lane = threadIdx.x;
  const int g = lane >> 4, ln = lane & 15;
  const int b  = blockIdx.x >> 8;
  const int i0 = (blockIdx.x & 255) << 4;
  const int chalf = blockIdx.y;

  // q B-tile [K=c(32)][N=i(16)]: lane N=ln, K-pairs contiguous in qT[b][i][c]
  U16 qb;
  {
    const unsigned int* qrow = (const unsigned int*)(qT + ((size_t)b * KN + i0 + ln) * KC8);
#pragma unroll
    for (int v = 0; v < 8; ++v) qb.u[v] = qrow[(g * 16 + 2 * v) >> 1];
  }

  v8f zero = {};
  v8f oacc[8];                       // O^T acc: vgpr r -> c-row r+8g, col i=ln
#pragma unroll
  for (int t = 0; t < 8; ++t) oacc[t] = zero;
  float mold = -3.0e38f, lsum = 0.0f;   // per-query (per-lane) running stats

  const _Float16* kTb = kT + (size_t)b * KN * KC8;
  const _Float16* vb  = vK + ((size_t)b * KC + chalf * 128 + ln) * KN;

#pragma unroll 1
  for (int jb = 0; jb < KN; jb += 128) {
    // --- scores: 8 S^T tiles (4 chunks x 2 j-halves), each 16 keys ---
    v8f st[8];
#pragma unroll
    for (int ch = 0; ch < 4; ++ch)
#pragma unroll
      for (int jh = 0; jh < 2; ++jh) {
        U16 ka;   // A[j-row, c]: lane M=ln -> key row, K-pairs contiguous in kT
        const unsigned int* krow =
            (const unsigned int*)(kTb + (size_t)(jb + ch * 32 + jh * 16 + ln) * KC8);
#pragma unroll
        for (int v = 0; v < 8; ++v) {
          int K = g * 16 + 2 * v;
          ka.u[v] = krow[K >> 1];
        }
        st[2 * ch + jh] = __builtin_amdgcn_wmma_f32_16x16x32_f16(
            false, ka.h, false, qb.h, (short)0, zero, false, false);
      }

    // --- online softmax over this 128-key block, per query column ---
    float mloc = st[0][0];
#pragma unroll
    for (int s = 0; s < 8; ++s)
#pragma unroll
      for (int r = 0; r < 8; ++r) mloc = fmaxf(mloc, st[s][r]);
    mloc = fmaxf(mloc, __shfl_xor(mloc, 16));      // combine lane pair halves
    float mnew  = fmaxf(mold, mloc);
    float scale = __expf(mold - mnew);
    mold = mnew;
    float ssum = 0.0f;
#pragma unroll
    for (int s = 0; s < 8; ++s)
#pragma unroll
      for (int r = 0; r < 8; ++r) {
        float p = __expf(st[s][r] - mnew);
        st[s][r] = p;                              // S becomes P (unnormalized)
        ssum += p;
      }
    ssum += __shfl_xor(ssum, 16);
    lsum = lsum * scale + ssum;
#pragma unroll
    for (int t = 0; t < 8; ++t)
#pragma unroll
      for (int r = 0; r < 8; ++r) oacc[t][r] *= scale;   // scale uniform/lane

    // --- accumulate: per 32-key chunk, build P^T B-tile then 8 WMMAs ---
#pragma unroll
    for (int ch = 0; ch < 4; ++ch) {
      // lane group g needs key offsets g*16..g*16+15 of this chunk for its
      // column i=ln; own regs give offsets r+8g, xor-16 partner the rest.
      float lo[8], hi8[8];
#pragma unroll
      for (int r = 0; r < 8; ++r) {
        float o0 = __shfl_xor(st[2 * ch][r], 16);      // partner's half-0 tile
        float o1 = __shfl_xor(st[2 * ch + 1][r], 16);  // partner's half-1 tile
        lo[r]  = g ? o1 : st[2 * ch][r];               // key offsets 0..7
        hi8[r] = g ? st[2 * ch + 1][r] : o0;           // key offsets 8..15
      }
      U16 pb;
#pragma unroll
      for (int j = 0; j < 8; ++j) {
        pb.h[j]     = (_Float16)lo[j];
        pb.h[8 + j] = (_Float16)hi8[j];
      }
#pragma unroll
      for (int t = 0; t < 8; ++t) {
        U16 va;   // A[c-row, n]: lane M=ln -> channel row, n-pairs contiguous
        const unsigned int* vrow =
            (const unsigned int*)(vb + (size_t)t * 16 * KN + jb + ch * 32);
#pragma unroll
        for (int v = 0; v < 8; ++v) {
          int K = g * 16 + 2 * v;
          va.u[v] = vrow[K >> 1];
        }
        oacc[t] = __builtin_amdgcn_wmma_f32_16x16x32_f16(
            false, va.h, false, pb.h, (short)0, oacc[t], false, false);
      }
    }
  }

  // --- epilogue: out = gamma * O/l + x ---
  float gm  = gamma[0];
  float inv = 1.0f / lsum;
#pragma unroll
  for (int t = 0; t < 8; ++t)
#pragma unroll
    for (int r = 0; r < 8; ++r) {
      int c = chalf * 128 + t * 16 + r + 8 * g;
      size_t idx = ((size_t)b * KC + c) * KN + i0 + ln;
      out[idx] = gm * oacc[t][r] * inv + x[idx];
    }
}

// ---------------------------------------------------------------------------
extern "C" void kernel_launch(void* const* d_in, const int* in_sizes, int n_in,
                              void* d_out, int out_size, void* d_ws, size_t ws_size,
                              hipStream_t stream) {
  (void)in_sizes; (void)n_in; (void)out_size; (void)ws_size;
  const float* x      = (const float*)d_in[0];
  const float* q_dw_w = (const float*)d_in[1];
  const float* q_dw_b = (const float*)d_in[2];
  const float* q_pw_w = (const float*)d_in[3];
  const float* q_pw_b = (const float*)d_in[4];
  const float* k_dw_w = (const float*)d_in[5];
  const float* k_dw_b = (const float*)d_in[6];
  const float* k_pw_w = (const float*)d_in[7];
  const float* k_pw_b = (const float*)d_in[8];
  const float* v_dw_w = (const float*)d_in[9];
  const float* v_dw_b = (const float*)d_in[10];
  const float* v_pw_w = (const float*)d_in[11];
  const float* v_pw_b = (const float*)d_in[12];
  const float* gamma  = (const float*)d_in[13];
  float* out = (float*)d_out;

  char* p = (char*)d_ws;
  auto take = [&](size_t bytes) { char* r = p; p += (bytes + 255) & ~(size_t)255; return r; };
  _Float16* wq16 = (_Float16*)take((size_t)KC8 * KC * 2);
  _Float16* wk16 = (_Float16*)take((size_t)KC8 * KC * 2);
  _Float16* wv16 = (_Float16*)take((size_t)KC * KC * 2);
  float* bq = (float*)take(KC8 * 4);
  float* bk = (float*)take(KC8 * 4);
  float* bv = (float*)take(KC * 4);
  _Float16* qT = (_Float16*)take((size_t)KB * KN * KC8 * 2);
  _Float16* kT = (_Float16*)take((size_t)KB * KN * KC8 * 2);
  _Float16* vK = (_Float16*)take((size_t)KB * KC * KN * 2);

  fold_weights_kernel<<<dim3(2 * KC8 + KC), dim3(KC), 0, stream>>>(
      q_dw_w, q_dw_b, q_pw_w, q_pw_b, k_dw_w, k_dw_b, k_pw_w, k_pw_b,
      v_dw_w, v_dw_b, v_pw_w, v_pw_b, wq16, wk16, wv16, bq, bk, bv);

  proj_qkv_kernel<<<dim3(KB * (KN / 16)), dim3(32), 0, stream>>>(
      x, wq16, wk16, wv16, bq, bk, bv, qT, kT, vK);

  attn_kernel<<<dim3(KB * (KN / 16), 2), dim3(32), 0, stream>>>(
      x, qT, kT, vK, gamma, out);
}
